// HyperbolicGraphConv_47596827574586
// MI455X (gfx1250) — matrix operand order repaired and checked by reading, not compiled, not run on hardware
//
#include <hip/hip_runtime.h>
#include <hip/hip_bf16.h>
#include <math.h>

#define N_NODES 50000
#define N_EDGES 800000
#define DIM     128
#define EPS_F   1e-12f

typedef __attribute__((ext_vector_type(16))) _Float16 v16h;
typedef __attribute__((ext_vector_type(8)))  _Float16 v8h;
typedef __attribute__((ext_vector_type(4)))  _Float16 v4h;
typedef __attribute__((ext_vector_type(8)))  float    v8f;

__device__ __forceinline__ float wave_reduce_sum(float v) {
#pragma unroll
  for (int m = 16; m >= 1; m >>= 1) v += __shfl_xor(v, m, 32);
  return v;
}

// ---------------- Kernel 1: tangent16 = (f16) log_map(0, x, c) -----------
// One wave (32 lanes) per node; each lane owns 4 contiguous floats.
// Stores the tangent row directly as packed f16 (same rounding the GEMM
// would apply, but done once -> no converts inside the WMMA loop).
__global__ __launch_bounds__(256)
void k_logmap(const float* __restrict__ x, const float* __restrict__ cptr,
              _Float16* __restrict__ tangent16) {
  const int wave = threadIdx.x >> 5, lane = threadIdx.x & 31;
  const int node = blockIdx.x * 8 + wave;
  if (node >= N_NODES) return;
  const float c  = *cptr;
  const float sc = sqrtf(c);
  float4 v = ((const float4*)(x + (size_t)node * DIM))[lane];
  float ss = v.x * v.x + v.y * v.y + v.z * v.z + v.w * v.w;
  ss = wave_reduce_sum(ss);
  const float norm = sqrtf(ss);
  const float vn   = fmaxf(norm, EPS_F);
  const float scale = (2.0f / sc) * atanhf(sc * vn) / vn;
  v4h o;
  o[0] = (_Float16)(scale * v.x);
  o[1] = (_Float16)(scale * v.y);
  o[2] = (_Float16)(scale * v.z);
  o[3] = (_Float16)(scale * v.w);
  ((v4h*)(tangent16 + (size_t)node * DIM))[lane] = o;
}

// ---------------- Kernel 1b: one-shot W -> f16 (64 KB, trivial) ----------
__global__ __launch_bounds__(256)
void k_cvt_w(const float* __restrict__ W, _Float16* __restrict__ W16) {
  const int i = blockIdx.x * blockDim.x + threadIdx.x;  // 0 .. 16383
  W16[i] = (_Float16)W[i];
}

// ---------------- Kernel 2: H = T @ W^T + b  via V_WMMA_F32_16X16X32_F16 --
// One wave computes a 16x16 tile of H. Wave w of the block handles N-tile w.
// grid = 3125 blocks (50000/16), block = 256 threads = 8 waves = all 8 N-tiles.
// Per K-step: 4x global_load_b128 (f16) + 1x v_wmma, no converts.
__global__ __launch_bounds__(256)
void k_gemm_wmma(const _Float16* __restrict__ T16, const _Float16* __restrict__ W16,
                 const float* __restrict__ bias, float* __restrict__ H) {
  const int lane = threadIdx.x & 31;
  const int wave = threadIdx.x >> 5;       // N-tile index 0..7
  const int m0 = blockIdx.x * 16;
  const int n0 = wave * 16;
  const int hsel = lane >> 4;              // lane half: 0 or 1
  const int r = lane & 15;                 // A: row m offset; B/D: col n offset

  const _Float16* arow = T16 + (size_t)(m0 + r) * DIM;  // A row (this lane's M)
  const _Float16* brow = W16 + (size_t)(n0 + r) * DIM;  // B col n0+r == W row (contig K)

  v8f acc = {};
#pragma unroll
  for (int kb = 0; kb < DIM; kb += 32) {
    // A fragment (16-bit A 16x32 layout): two contiguous 8-half runs
    //   halves 0..7  -> K = kb +      8*hsel .. +7
    //   halves 8..15 -> K = kb + 16 + 8*hsel .. +7
    const v8h a0 = *(const v8h*)(arow + kb + 8 * hsel);
    const v8h a1 = *(const v8h*)(arow + kb + 16 + 8 * hsel);
    const v16h a = __builtin_shufflevector(a0, a1,
        0, 1, 2, 3, 4, 5, 6, 7, 8, 9, 10, 11, 12, 13, 14, 15);
    // B fragment (32x16 layout): same K order as A (kb + 8*hsel, kb + 16 + 8*hsel)
    const v8h b0 = *(const v8h*)(brow + kb + 8 * hsel);
    const v8h b1 = *(const v8h*)(brow + kb + 16 + 8 * hsel);
    const v16h bm = __builtin_shufflevector(b0, b1,
        0, 1, 2, 3, 4, 5, 6, 7, 8, 9, 10, 11, 12, 13, 14, 15);

    acc = __builtin_amdgcn_wmma_f32_16x16x32_f16(
        /*neg_a=*/false, a, /*neg_b=*/false, bm,
        /*c_mod=*/(short)0, acc, /*reuse_a=*/false, /*reuse_b=*/false);
  }
  // D layout: lane n = r; VGPR v holds M = v + 8*hsel
  const float bv = bias[n0 + r];
#pragma unroll
  for (int v = 0; v < 8; ++v) {
    const int m = m0 + v + 8 * hsel;
    H[(size_t)m * DIM + n0 + r] = acc[v] + bv;
  }
}

// ---------------- Kernel 3a: zero accumulators --------------------------
__global__ __launch_bounds__(256)
void k_zero(float* __restrict__ p, int n) {
  const int i = blockIdx.x * blockDim.x + threadIdx.x;
  if (i < n) p[i] = 0.0f;
}

// ---------------- Kernel 3b: edge scatter (copy_u + sum, deg count) -----
// One wave per edge; lane owns 4 floats of the 128-wide message.
// h / summed working sets (25.6 MB each) stay resident in the 192 MB L2.
__global__ __launch_bounds__(256)
void k_scatter(const float* __restrict__ H, const int* __restrict__ src,
               const int* __restrict__ dst, float* __restrict__ summed,
               float* __restrict__ deg) {
  const int wave = threadIdx.x >> 5, lane = threadIdx.x & 31;
  const int e = blockIdx.x * 8 + wave;
  if (e >= N_EDGES) return;
  const int s = src[e], d = dst[e];
  const float4 v = ((const float4*)(H + (size_t)s * DIM))[lane];
  float* out = summed + (size_t)d * DIM + lane * 4;
  atomicAdd(out + 0, v.x);
  atomicAdd(out + 1, v.y);
  atomicAdd(out + 2, v.z);
  atomicAdd(out + 3, v.w);
  if (lane == 0) atomicAdd(deg + d, 1.0f);
}

// ---------------- Kernel 4: out = exp_map(0, summed/deg, c), in place ----
__global__ __launch_bounds__(256)
void k_expmap(const float* __restrict__ deg, const float* __restrict__ cptr,
              float* __restrict__ out) {
  const int wave = threadIdx.x >> 5, lane = threadIdx.x & 31;
  const int node = blockIdx.x * 8 + wave;
  if (node >= N_NODES) return;
  const float c  = *cptr;
  const float sc = sqrtf(c);
  const float inv_d = 1.0f / fmaxf(deg[node], 1.0f);
  float4 v = ((const float4*)(out + (size_t)node * DIM))[lane];
  v.x *= inv_d; v.y *= inv_d; v.z *= inv_d; v.w *= inv_d;
  float ss = v.x * v.x + v.y * v.y + v.z * v.z + v.w * v.w;
  ss = wave_reduce_sum(ss);
  const float norm = sqrtf(ss);
  const float vn   = fmaxf(norm, EPS_F);
  const float scale = tanhf(sc * vn * 0.5f) / (sc * vn);
  float4 o = make_float4(scale * v.x, scale * v.y, scale * v.z, scale * v.w);
  ((float4*)(out + (size_t)node * DIM))[lane] = o;
}

// ---------------- launcher ----------------------------------------------
extern "C" void kernel_launch(void* const* d_in, const int* in_sizes, int n_in,
                              void* d_out, int out_size, void* d_ws, size_t ws_size,
                              hipStream_t stream) {
  (void)in_sizes; (void)n_in; (void)out_size; (void)ws_size;
  const float* x    = (const float*)d_in[0];
  const float* W    = (const float*)d_in[1];
  const float* bias = (const float*)d_in[2];
  const int*   src  = (const int*)d_in[3];
  const int*   dst  = (const int*)d_in[4];
  const float* curv = (const float*)d_in[5];
  float* out = (float*)d_out;

  // Workspace layout (d_ws is >=256B aligned):
  //   H        : N_NODES*DIM f32   (25.6 MB)
  //   tangent16: N_NODES*DIM f16   (12.8 MB)
  //   W16      : DIM*DIM f16       (32 KB)
  //   deg      : N_NODES f32       (200 KB)
  char* wsb = (char*)d_ws;
  float*    H   = (float*)wsb;
  _Float16* T16 = (_Float16*)(wsb + (size_t)N_NODES * DIM * sizeof(float));
  _Float16* W16 = T16 + (size_t)N_NODES * DIM;
  float*    deg = (float*)(W16 + (size_t)DIM * DIM);

  // d_out doubles as the atomic accumulator; zero it + deg every call.
  {
    const int n = N_NODES * DIM;
    k_zero<<<(n + 255) / 256, 256, 0, stream>>>(out, n);
    k_zero<<<(N_NODES + 255) / 256, 256, 0, stream>>>(deg, N_NODES);
  }
  k_cvt_w   <<<(DIM * DIM) / 256,  256, 0, stream>>>(W, W16);
  k_logmap  <<<(N_NODES + 7) / 8,  256, 0, stream>>>(x, curv, T16);
  k_gemm_wmma<<<N_NODES / 16,      256, 0, stream>>>(T16, W16, bias, H);
  k_scatter <<<(N_EDGES + 7) / 8,  256, 0, stream>>>(H, src, dst, out, deg);
  k_expmap  <<<(N_NODES + 7) / 8,  256, 0, stream>>>(deg, curv, out);
}
